// DGCNN_68066641707931
// MI455X (gfx1250) — compile-verified, matrix operand order checked by reading, not run on hardware
//
#include <hip/hip_runtime.h>
#include <hip/hip_bf16.h>
#include <math.h>

// ---------------------------------------------------------------------------
// DGCNN forward for MI455X (gfx1250, wave32, WMMA).
// f16 operands, f32 accumulation via v_wmma_f32_16x16x32_f16.
// Weight tiles live in LDS (transposed, f16), filled with async global->LDS.
// Neighbor loop unrolled by 2 so each LDS B-fragment feeds two WMMAs.
// ---------------------------------------------------------------------------

typedef __attribute__((ext_vector_type(16))) _Float16 v16h;
typedef __attribute__((ext_vector_type(8)))  _Float16 v8h;
typedef __attribute__((ext_vector_type(8)))  float    v8f;

#define NPTS 32768
#define KNB  20
#define NEG_BIG (-3.0e38f)

// f16 weight blob layout (halves), transposed to [out][in]:
#define W11T_OFF 0                      // [64][32]  (k>=12 zero padded)
#define W12T_OFF (W11T_OFF + 64 * 32)   // [64][64]
#define W13T_OFF (W12T_OFF + 64 * 64)   // [128][64]
#define WGT1_HALVES (W13T_OFF + 128 * 64)          // 14336 halves = 28672 B
#define W21T_OFF 0                      // [128][256]
#define W22T_OFF (W21T_OFF + 128 * 256) // [128][128]
#define W23T_OFF (W22T_OFF + 128 * 128) // [256][128]
#define WGT2_HALVES (W23T_OFF + 256 * 128)         // 81920 halves = 163840 B

__device__ __forceinline__ v8f wmma_f16(v16h a, v16h b, v8f c) {
  return __builtin_amdgcn_wmma_f32_16x16x32_f16(
      false, a, false, b, (short)0, c, false, false);
}

__device__ __forceinline__ v16h ld_frag(const _Float16* p) {
  v8h lo = *(const v8h*)p;
  v8h hi = *(const v8h*)(p + 8);
  v16h r;
#pragma unroll
  for (int i = 0; i < 8; ++i) { r[i] = lo[i]; r[i + 8] = hi[i]; }
  return r;
}

// Wave-local LDS fence (LDS ops are in-order per wave; DScnt==0 => visible).
__device__ __forceinline__ void lds_wave_sync() {
  asm volatile("s_wait_dscnt 0" ::: "memory");
}

// CDNA5 async global->LDS copy (ASYNCcnt-tracked, no VGPR round trip).
__device__ __forceinline__ void async_copy_b128(unsigned lds_off, const void* gaddr) {
  asm volatile("global_load_async_to_lds_b128 %0, %1, off"
               :: "v"(lds_off), "v"(gaddr) : "memory");
}
__device__ __forceinline__ void async_wait() {
  asm volatile("s_wait_asynccnt 0" ::: "memory");
}

__device__ __forceinline__ v8f bias_ctile(const float* __restrict__ bias, int col) {
  float bv = bias[col];
  v8f c;
#pragma unroll
  for (int r = 0; r < 8; ++r) c[r] = bv;
  return c;
}

// ---------------------------------------------------------------------------
// Prep: convert all conv weights to transposed f16 blobs in workspace.
// ---------------------------------------------------------------------------
__global__ __launch_bounds__(256) void prep_weights_kernel(
    const float* __restrict__ w11, const float* __restrict__ w12,
    const float* __restrict__ w13, const float* __restrict__ w21,
    const float* __restrict__ w22, const float* __restrict__ w23,
    _Float16* __restrict__ g1, _Float16* __restrict__ g2) {
  const int t = threadIdx.x;
  for (int i = t; i < 64 * 32; i += 256) {
    int n = i >> 5, k = i & 31;
    g1[W11T_OFF + i] = (k < 12) ? (_Float16)w11[k * 64 + n] : (_Float16)0.0f;
  }
  for (int i = t; i < 64 * 64; i += 256) {
    int n = i >> 6, k = i & 63;
    g1[W12T_OFF + i] = (_Float16)w12[k * 64 + n];
  }
  for (int i = t; i < 128 * 64; i += 256) {
    int n = i >> 6, k = i & 63;
    g1[W13T_OFF + i] = (_Float16)w13[k * 128 + n];
  }
  for (int i = t; i < 128 * 256; i += 256) {
    int n = i >> 8, k = i & 255;
    g2[W21T_OFF + i] = (_Float16)w21[k * 128 + n];
  }
  for (int i = t; i < 128 * 128; i += 256) {
    int n = i >> 7, k = i & 127;
    g2[W22T_OFF + i] = (_Float16)w22[k * 128 + n];
  }
  for (int i = t; i < 256 * 128; i += 256) {
    int n = i >> 7, k = i & 127;
    g2[W23T_OFF + i] = (_Float16)w23[k * 256 + n];
  }
}

// ---------------------------------------------------------------------------
// EdgeConv1: concat(x[j],x[i]) (12) -> 64 -> 64 -> 128, max over K=20.
// 4 waves/block, one 16-point tile per wave, neighbor loop unrolled x2.
// 36 KB static LDS: 28 KB weights + 4 waves x 4 staging tiles (16x64 f16).
// ---------------------------------------------------------------------------
__global__ __launch_bounds__(128) void edgeconv1_kernel(
    const float* __restrict__ x, const int* __restrict__ idx,
    const _Float16* __restrict__ wgt1,
    const float* __restrict__ b11, const float* __restrict__ b12,
    const float* __restrict__ b13, _Float16* __restrict__ h1) {
  __shared__ _Float16 wlds[WGT1_HALVES + 4 * 4 * 16 * 64];

  const int tid = threadIdx.x;
  // Async fill of the f16 weight blob into LDS.
  {
    const char* src = (const char*)wgt1;
    unsigned dst = (unsigned)(uintptr_t)&wlds[0];
    for (int off = tid * 16; off < (int)(WGT1_HALVES * 2); off += 128 * 16)
      async_copy_b128(dst + off, src + off);
    async_wait();
  }
  __syncthreads();

  const _Float16* w11t = wlds + W11T_OFF;
  const _Float16* w12t = wlds + W12T_OFF;
  const _Float16* w13t = wlds + W13T_OFF;

  const int lane  = tid & 31;
  const int wv    = tid >> 5;
  const int tile  = blockIdx.x * 4 + wv;
  const int row   = lane & 15;
  const int pt    = tile * 16 + row;
  const int colc  = lane & 15;
  const int rbase = (lane >> 4) * 8;
  const int khalf = (lane >> 4) << 4;

  _Float16* s0a = wlds + WGT1_HALVES + wv * 4 * (16 * 64);
  _Float16* s0b = s0a + 16 * 64;
  _Float16* s1a = s0b + 16 * 64;
  _Float16* s1b = s1a + 16 * 64;

  float vmax[8][8];
#pragma unroll
  for (int nt = 0; nt < 8; ++nt)
#pragma unroll
    for (int r = 0; r < 8; ++r) vmax[nt][r] = NEG_BIG;

  for (int k = 0; k < KNB; k += 2) {
    // Build A fragments for two neighbors (16 edges x 32, real K=12).
    v16h a0, a1;
#pragma unroll
    for (int i = 0; i < 16; ++i) { a0[i] = (_Float16)0.0f; a1[i] = (_Float16)0.0f; }
    if (lane < 16) {
      int j0 = idx[pt * KNB + k];
      int j1 = idx[pt * KNB + k + 1];
      const float* xj0 = x + j0 * 6;
      const float* xj1 = x + j1 * 6;
      const float* xi  = x + pt * 6;
#pragma unroll
      for (int f = 0; f < 6; ++f) {
        a0[f] = (_Float16)xj0[f];
        a1[f] = (_Float16)xj1[f];
        _Float16 c = (_Float16)xi[f];
        a0[6 + f] = c;
        a1[6 + f] = c;
      }
    }

    // GEMM1: (12->64) + bias + relu; one B fragment feeds both neighbors.
#pragma unroll
    for (int nt = 0; nt < 4; ++nt) {
      v8f c0 = bias_ctile(b11, nt * 16 + colc);
      v8f c1 = c0;
      v16h b = ld_frag(w11t + (nt * 16 + colc) * 32 + khalf);
      c0 = wmma_f16(a0, b, c0);
      c1 = wmma_f16(a1, b, c1);
#pragma unroll
      for (int r = 0; r < 8; ++r) {
        float v0 = c0[r], v1 = c1[r];
        s0a[(rbase + r) * 64 + nt * 16 + colc] = (_Float16)(v0 > 0.0f ? v0 : 0.0f);
        s0b[(rbase + r) * 64 + nt * 16 + colc] = (_Float16)(v1 > 0.0f ? v1 : 0.0f);
      }
    }
    lds_wave_sync();

    // GEMM2: (64->64) + bias + relu
    {
      v16h p0 = ld_frag(s0a + row * 64 + khalf);
      v16h p1 = ld_frag(s0a + row * 64 + 32 + khalf);
      v16h q0 = ld_frag(s0b + row * 64 + khalf);
      v16h q1 = ld_frag(s0b + row * 64 + 32 + khalf);
#pragma unroll
      for (int nt = 0; nt < 4; ++nt) {
        v8f c0 = bias_ctile(b12, nt * 16 + colc);
        v8f c1 = c0;
        v16h bA = ld_frag(w12t + (nt * 16 + colc) * 64 + khalf);
        v16h bB = ld_frag(w12t + (nt * 16 + colc) * 64 + 32 + khalf);
        c0 = wmma_f16(p0, bA, c0); c1 = wmma_f16(q0, bA, c1);
        c0 = wmma_f16(p1, bB, c0); c1 = wmma_f16(q1, bB, c1);
#pragma unroll
        for (int r = 0; r < 8; ++r) {
          float v0 = c0[r], v1 = c1[r];
          s1a[(rbase + r) * 64 + nt * 16 + colc] = (_Float16)(v0 > 0.0f ? v0 : 0.0f);
          s1b[(rbase + r) * 64 + nt * 16 + colc] = (_Float16)(v1 > 0.0f ? v1 : 0.0f);
        }
      }
    }
    lds_wave_sync();

    // GEMM3: (64->128) + bias, fold into running max.
    {
      v16h p0 = ld_frag(s1a + row * 64 + khalf);
      v16h p1 = ld_frag(s1a + row * 64 + 32 + khalf);
      v16h q0 = ld_frag(s1b + row * 64 + khalf);
      v16h q1 = ld_frag(s1b + row * 64 + 32 + khalf);
#pragma unroll
      for (int nt = 0; nt < 8; ++nt) {
        v8f c0 = bias_ctile(b13, nt * 16 + colc);
        v8f c1 = c0;
        v16h bA = ld_frag(w13t + (nt * 16 + colc) * 64 + khalf);
        v16h bB = ld_frag(w13t + (nt * 16 + colc) * 64 + 32 + khalf);
        c0 = wmma_f16(p0, bA, c0); c1 = wmma_f16(q0, bA, c1);
        c0 = wmma_f16(p1, bB, c0); c1 = wmma_f16(q1, bB, c1);
#pragma unroll
        for (int r = 0; r < 8; ++r)
          vmax[nt][r] = fmaxf(vmax[nt][r], fmaxf(c0[r], c1[r]));
      }
    }
    lds_wave_sync();
  }

#pragma unroll
  for (int nt = 0; nt < 8; ++nt)
#pragma unroll
    for (int r = 0; r < 8; ++r)
      h1[(size_t)(tile * 16 + rbase + r) * 128 + nt * 16 + colc] =
          (_Float16)vmax[nt][r];
}

// ---------------------------------------------------------------------------
// EdgeConv2: concat(h1[j],h1[i]) (256) -> 128 -> 128 -> 256, max over K=20.
// 4 waves/block; 224 KB dynamic LDS (160 KB weights + 64 KB staging) — only
// possible because the MI455X WGP has 320 KB LDS. Neighbor loop unrolled x2.
// ---------------------------------------------------------------------------
__global__ __launch_bounds__(128) void edgeconv2_kernel(
    const _Float16* __restrict__ h1, const int* __restrict__ idx,
    const _Float16* __restrict__ wgt2,
    const float* __restrict__ b21, const float* __restrict__ b22,
    const float* __restrict__ b23, _Float16* __restrict__ h2) {
  extern __shared__ _Float16 sm[];
  const int tid = threadIdx.x;

  // Async fill of the 160 KB f16 weight blob into LDS.
  {
    const char* src = (const char*)wgt2;
    unsigned dst = (unsigned)(uintptr_t)&sm[0];
    for (int off = tid * 16; off < (int)(WGT2_HALVES * 2); off += 128 * 16)
      async_copy_b128(dst + off, src + off);
    async_wait();
  }
  __syncthreads();

  const _Float16* w21t = sm + W21T_OFF;
  const _Float16* w22t = sm + W22T_OFF;
  const _Float16* w23t = sm + W23T_OFF;
  _Float16* stg = sm + WGT2_HALVES;

  const int lane  = tid & 31;
  const int wv    = tid >> 5;
  const int tile  = blockIdx.x * 4 + wv;
  const int row   = lane & 15;
  const int pt    = tile * 16 + row;
  const int colc  = lane & 15;
  const int rbase = (lane >> 4) * 8;
  const int khalf = (lane >> 4) << 4;

  _Float16* s0a = stg + wv * 4 * (16 * 128);
  _Float16* s0b = s0a + 16 * 128;
  _Float16* s1a = s0b + 16 * 128;
  _Float16* s1b = s1a + 16 * 128;

  float vmax[16][8];
#pragma unroll
  for (int nt = 0; nt < 16; ++nt)
#pragma unroll
    for (int r = 0; r < 8; ++r) vmax[nt][r] = NEG_BIG;

  // Center-point A fragments (cols 128..255 of the concat) are loop-invariant.
  v16h ac[4];
#pragma unroll
  for (int t = 0; t < 4; ++t)
    ac[t] = ld_frag(h1 + (size_t)pt * 128 + t * 32 + khalf);

  for (int k = 0; k < KNB; k += 2) {
    int j0 = idx[pt * KNB + k];
    int j1 = idx[pt * KNB + k + 1];
    if (k + 2 < KNB) {
      int jn = idx[pt * KNB + k + 2];
      __builtin_prefetch(h1 + (size_t)jn * 128, 0, 1);  // global_prefetch_b8
    }

    v16h an0[4], an1[4];
#pragma unroll
    for (int t = 0; t < 4; ++t) {
      an0[t] = ld_frag(h1 + (size_t)j0 * 128 + t * 32 + khalf);
      an1[t] = ld_frag(h1 + (size_t)j1 * 128 + t * 32 + khalf);
    }

    // GEMM1: (256->128) + bias + relu; B fragments shared by both neighbors.
#pragma unroll
    for (int nt = 0; nt < 8; ++nt) {
      v8f c0 = bias_ctile(b21, nt * 16 + colc);
      v8f c1 = c0;
      const _Float16* wb = w21t + (nt * 16 + colc) * 256;
#pragma unroll
      for (int kt = 0; kt < 4; ++kt) {
        v16h b = ld_frag(wb + kt * 32 + khalf);
        c0 = wmma_f16(an0[kt], b, c0);
        c1 = wmma_f16(an1[kt], b, c1);
      }
#pragma unroll
      for (int kt = 0; kt < 4; ++kt) {
        v16h b = ld_frag(wb + (kt + 4) * 32 + khalf);
        c0 = wmma_f16(ac[kt], b, c0);
        c1 = wmma_f16(ac[kt], b, c1);
      }
#pragma unroll
      for (int r = 0; r < 8; ++r) {
        float v0 = c0[r], v1 = c1[r];
        s0a[(rbase + r) * 128 + nt * 16 + colc] = (_Float16)(v0 > 0.0f ? v0 : 0.0f);
        s0b[(rbase + r) * 128 + nt * 16 + colc] = (_Float16)(v1 > 0.0f ? v1 : 0.0f);
      }
    }
    lds_wave_sync();

    // GEMM2: (128->128) + bias + relu
    {
      v16h p[4], q[4];
#pragma unroll
      for (int t = 0; t < 4; ++t) {
        p[t] = ld_frag(s0a + row * 128 + t * 32 + khalf);
        q[t] = ld_frag(s0b + row * 128 + t * 32 + khalf);
      }
#pragma unroll
      for (int nt = 0; nt < 8; ++nt) {
        v8f c0 = bias_ctile(b22, nt * 16 + colc);
        v8f c1 = c0;
        const _Float16* wb = w22t + (nt * 16 + colc) * 128;
#pragma unroll
        for (int kt = 0; kt < 4; ++kt) {
          v16h b = ld_frag(wb + kt * 32 + khalf);
          c0 = wmma_f16(p[kt], b, c0);
          c1 = wmma_f16(q[kt], b, c1);
        }
#pragma unroll
        for (int r = 0; r < 8; ++r) {
          float v0 = c0[r], v1 = c1[r];
          s1a[(rbase + r) * 128 + nt * 16 + colc] = (_Float16)(v0 > 0.0f ? v0 : 0.0f);
          s1b[(rbase + r) * 128 + nt * 16 + colc] = (_Float16)(v1 > 0.0f ? v1 : 0.0f);
        }
      }
    }
    lds_wave_sync();

    // GEMM3: (128->256) + bias, fold into running max.
    {
      v16h p[4], q[4];
#pragma unroll
      for (int t = 0; t < 4; ++t) {
        p[t] = ld_frag(s1a + row * 128 + t * 32 + khalf);
        q[t] = ld_frag(s1b + row * 128 + t * 32 + khalf);
      }
#pragma unroll
      for (int nt = 0; nt < 16; ++nt) {
        v8f c0 = bias_ctile(b23, nt * 16 + colc);
        v8f c1 = c0;
        const _Float16* wb = w23t + (nt * 16 + colc) * 128;
#pragma unroll
        for (int kt = 0; kt < 4; ++kt) {
          v16h b = ld_frag(wb + kt * 32 + khalf);
          c0 = wmma_f16(p[kt], b, c0);
          c1 = wmma_f16(q[kt], b, c1);
        }
#pragma unroll
        for (int r = 0; r < 8; ++r)
          vmax[nt][r] = fmaxf(vmax[nt][r], fmaxf(c0[r], c1[r]));
      }
    }
    lds_wave_sync();
  }

#pragma unroll
  for (int nt = 0; nt < 16; ++nt)
#pragma unroll
    for (int r = 0; r < 8; ++r)
      h2[(size_t)(tile * 16 + rbase + r) * 256 + nt * 16 + colc] =
          (_Float16)vmax[nt][r];
}

// ---------------------------------------------------------------------------
// segment_max over sorted batch ids.
// ---------------------------------------------------------------------------
__global__ __launch_bounds__(256) void segmax_kernel(
    const _Float16* __restrict__ h2, const int* __restrict__ batch,
    float* __restrict__ pooled) {
  const int b = blockIdx.x;
  const int c = threadIdx.x;

  int lo = 0, hi = NPTS;
  while (lo < hi) { int m = (lo + hi) >> 1; if (batch[m] <  b) lo = m + 1; else hi = m; }
  const int start = lo;
  lo = start; hi = NPTS;
  while (lo < hi) { int m = (lo + hi) >> 1; if (batch[m] <= b) lo = m + 1; else hi = m; }
  const int end = lo;

  float mx = NEG_BIG;
  for (int n = start; n < end; ++n)
    mx = fmaxf(mx, (float)h2[(size_t)n * 256 + c]);
  pooled[b * 256 + c] = mx;
}

// ---------------------------------------------------------------------------
// Head: relu(pooled@wl1+bl1) @ wl2 + bl2 -> log_softmax.
// ---------------------------------------------------------------------------
__global__ __launch_bounds__(256) void head_kernel(
    const float* __restrict__ pooled,
    const float* __restrict__ wl1, const float* __restrict__ bl1,
    const float* __restrict__ wl2, const float* __restrict__ bl2,
    float* __restrict__ out) {
  __shared__ float hbuf[8 * 256];
  __shared__ float lg[8 * 10];

  const int t = threadIdx.x;
  {
    float acc[8];
#pragma unroll
    for (int b = 0; b < 8; ++b) acc[b] = bl1[t];
    for (int i = 0; i < 256; ++i) {
      float w = wl1[i * 256 + t];
#pragma unroll
      for (int b = 0; b < 8; ++b) acc[b] += pooled[b * 256 + i] * w;
    }
#pragma unroll
    for (int b = 0; b < 8; ++b) hbuf[b * 256 + t] = acc[b] > 0.0f ? acc[b] : 0.0f;
  }
  __syncthreads();

  if (t < 80) {
    int b = t / 10, c = t % 10;
    float acc = bl2[c];
    for (int i = 0; i < 256; ++i) acc += hbuf[b * 256 + i] * wl2[i * 10 + c];
    lg[b * 10 + c] = acc;
  }
  __syncthreads();

  if (t < 8) {
    float mx = NEG_BIG;
    for (int c = 0; c < 10; ++c) mx = fmaxf(mx, lg[t * 10 + c]);
    float s = 0.0f;
    for (int c = 0; c < 10; ++c) s += expf(lg[t * 10 + c] - mx);
    float lse = mx + logf(s);
    for (int c = 0; c < 10; ++c) out[t * 10 + c] = lg[t * 10 + c] - lse;
  }
}

// ---------------------------------------------------------------------------
// Launch
// ---------------------------------------------------------------------------
extern "C" void kernel_launch(void* const* d_in, const int* in_sizes, int n_in,
                              void* d_out, int out_size, void* d_ws, size_t ws_size,
                              hipStream_t stream) {
  const float* x     = (const float*)d_in[0];
  const int*   idx   = (const int*)d_in[1];
  const int*   batch = (const int*)d_in[2];
  const float* w11 = (const float*)d_in[3],  *b11 = (const float*)d_in[4];
  const float* w12 = (const float*)d_in[5],  *b12 = (const float*)d_in[6];
  const float* w13 = (const float*)d_in[7],  *b13 = (const float*)d_in[8];
  const float* w21 = (const float*)d_in[9],  *b21 = (const float*)d_in[10];
  const float* w22 = (const float*)d_in[11], *b22 = (const float*)d_in[12];
  const float* w23 = (const float*)d_in[13], *b23 = (const float*)d_in[14];
  const float* wl1 = (const float*)d_in[15], *bl1 = (const float*)d_in[16];
  const float* wl2 = (const float*)d_in[17], *bl2 = (const float*)d_in[18];
  float* out = (float*)d_out;

  char* ws = (char*)d_ws;
  size_t off = 0;
  _Float16* h1  = (_Float16*)(ws + off); off += (size_t)NPTS * 128 * 2;  // 8 MiB
  _Float16* h2  = (_Float16*)(ws + off); off += (size_t)NPTS * 256 * 2;  // 16 MiB
  float* pooled = (float*)(ws + off);    off += 8 * 256 * 4;
  off = (off + 255) & ~(size_t)255;
  _Float16* g1  = (_Float16*)(ws + off); off += (size_t)WGT1_HALVES * 2;
  _Float16* g2  = (_Float16*)(ws + off); off += (size_t)WGT2_HALVES * 2;

  prep_weights_kernel<<<1, 256, 0, stream>>>(w11, w12, w13, w21, w22, w23, g1, g2);

  // EdgeConv1: 2048 tiles / 4 waves per block -> 512 blocks.
  edgeconv1_kernel<<<512, 128, 0, stream>>>(x, idx, g1, b11, b12, b13, h1);

  // EdgeConv2: 512 blocks; 224 KB dynamic LDS (needs WGP-mode 320 KB LDS).
  const size_t conv2_lds =
      (size_t)(WGT2_HALVES + 4 * 4 * 16 * 128) * sizeof(_Float16);
  (void)hipFuncSetAttribute((const void*)edgeconv2_kernel,
                            hipFuncAttributeMaxDynamicSharedMemorySize,
                            (int)conv2_lds);
  edgeconv2_kernel<<<512, 128, conv2_lds, stream>>>(h1, idx, g2, b21, b22, b23, h2);

  segmax_kernel<<<8, 256, 0, stream>>>(h2, batch, pooled);
  head_kernel<<<1, 256, 0, stream>>>(pooled, wl1, bl1, wl2, bl2, out);
}
